// CrossEncoder_88802743812416
// MI455X (gfx1250) — matrix operand order repaired and checked by reading, not compiled, not run on hardware
//
#include <hip/hip_runtime.h>
#include <math.h>

typedef __attribute__((ext_vector_type(2)))  float    v2f;
typedef __attribute__((ext_vector_type(8)))  float    v8f;
typedef __attribute__((ext_vector_type(16))) _Float16 v16h;

#define KNN 20
#define GN_EPS 1e-5f

#if __has_builtin(__builtin_amdgcn_sched_barrier)
#define SCHED_FENCE() __builtin_amdgcn_sched_barrier(0)
#else
#define SCHED_FENCE()
#endif

// ------------------------------------------------------------------
// K1: L2-normalize along C(=128) and transpose (B,C,N) -> (B,N,C).
// Optionally also writes the raw (un-normalized) transpose.
// ------------------------------------------------------------------
__global__ void __launch_bounds__(128)
k_norm_transpose(const float* __restrict__ src, float* __restrict__ dstN,
                 float* __restrict__ dstRaw, int Npts) {
  int gid = blockIdx.x * 128 + threadIdx.x;   // b*Npts + n
  int b = gid / Npts;
  int n = gid - b * Npts;
  const float* s = src + (size_t)b * 128 * Npts + n;
  float v[128];
  float ss = 0.f;
#pragma unroll
  for (int c = 0; c < 128; ++c) { float x = s[(size_t)c * Npts]; v[c] = x; ss += x * x; }
  float inv = 1.0f / fmaxf(sqrtf(ss), 1e-12f);
  float* dN = dstN + (size_t)gid * 128;
#pragma unroll
  for (int c = 0; c < 128; ++c) dN[c] = v[c] * inv;
  if (dstRaw) {
    float* dR = dstRaw + (size_t)gid * 128;
#pragma unroll
    for (int c = 0; c < 128; ++c) dR[c] = v[c];
  }
}

// ------------------------------------------------------------------
// K2: fused WMMA similarity (f32) + per-row top-20 + softmax + gather.
// One WG (8 waves) per 16 rows of N; each wave streams 32 of the 256
// column tiles (stride 8) and keeps a sorted top-20 list per row in LDS.
// sched_barrier(0) fences keep the 32 B-fragment global_load_b64 issued
// as one clause ahead of the back-to-back WMMA chain (incremental waits).
// ------------------------------------------------------------------
__global__ void __launch_bounds__(256)
k_simi_att(const float* __restrict__ xnT, const float* __restrict__ ynT,
           const float* __restrict__ y3T, const float* __restrict__ x3,
           float* __restrict__ att, int Npts, int Mpts) {
  const int b    = blockIdx.y;
  const int n0   = blockIdx.x * 16;
  const int tid  = threadIdx.x;
  const int wave = tid >> 5;
  const int lane = tid & 31;
  const int lmod = lane & 15;
  const int lhi  = lane >> 4;

  __shared__ float simL[8][16][16];
  __shared__ float lval[8][16][KNN];
  __shared__ int   lidx[8][16][KNN];
  __shared__ float wFin[16][KNN];
  __shared__ int   iFin[16][KNN];

  for (int i = tid; i < 8 * 16 * KNN; i += 256) {
    (&lval[0][0][0])[i] = -1e30f;
    (&lidx[0][0][0])[i] = 0;
  }
  __syncthreads();

  // A-fragments for all 32 K-steps: lane -> row n0+lmod, k = 4s + 2*lhi + j
  v2f a[32];
  {
    const float* ap = xnT + ((size_t)b * Npts + n0 + lmod) * 128 + 2 * lhi;
#pragma unroll
    for (int s = 0; s < 32; ++s) a[s] = *(const v2f*)(ap + 4 * s);
  }

  const int ntiles = Mpts >> 4;   // 256; every wave runs exactly 32 iterations
  for (int mt = wave; mt < ntiles; mt += 8) {
    const int m0 = mt * 16;
    v8f acc = {};
    const float* bp = ynT + ((size_t)b * Mpts + m0 + lmod) * 128 + 2 * lhi;
    v2f bfr[32];
    SCHED_FENCE();
#pragma unroll
    for (int s = 0; s < 32; ++s) bfr[s] = *(const v2f*)(bp + 4 * s);
    SCHED_FENCE();   // all 32 loads issue before any WMMA; waits become incremental
#pragma unroll
    for (int s = 0; s < 32; ++s)
      acc = __builtin_amdgcn_wmma_f32_16x16x4_f32(false, a[s], false, bfr[s],
                                                  (short)0, acc, false, false);
    SCHED_FENCE();
    // D layout: VGPR r -> row r + 8*lhi, col lmod
#pragma unroll
    for (int r = 0; r < 8; ++r) simL[wave][r + 8 * lhi][lmod] = acc[r];
    __syncthreads();
    if (lane < 16) {
      const int row = lane;
      float* LV = lval[wave][row];
      int*   LI = lidx[wave][row];
      for (int c = 0; c < 16; ++c) {
        float vv = simL[wave][row][c];
        if (vv > LV[KNN - 1]) {
          int p = KNN - 1;
          while (p > 0 && LV[p - 1] < vv) { LV[p] = LV[p - 1]; LI[p] = LI[p - 1]; --p; }
          LV[p] = vv; LI[p] = m0 + c;
        }
      }
    }
    __syncthreads();
  }

  // 8-way merge of per-wave lists, then softmax over the top-20 sims.
  if (tid < 16) {
    const int row = tid;
    int head[8] = {0, 0, 0, 0, 0, 0, 0, 0};
    for (int k = 0; k < KNN; ++k) {
      float bv = -1e38f; int bw = 0;
#pragma unroll
      for (int wv = 0; wv < 8; ++wv) {
        float vv = lval[wv][row][head[wv]];
        if (vv > bv) { bv = vv; bw = wv; }
      }
      wFin[row][k] = bv;
      iFin[row][k] = lidx[bw][row][head[bw]];
      head[bw]++;
    }
    float mx = wFin[row][0], sum = 0.f;
    for (int k = 0; k < KNN; ++k) { float e = expf(wFin[row][k] - mx); wFin[row][k] = e; sum += e; }
    float inv = 1.f / sum;
    for (int k = 0; k < KNN; ++k) wFin[row][k] *= inv;
  }
  __syncthreads();

  // agg[c] = sum_k w_k * y3T[idx_k][c] - x3[c,n]  (since sum w = 1)
#pragma unroll
  for (int it = 0; it < 8; ++it) {
    const int r = it * 2 + (tid >> 7);
    const int c = tid & 127;
    const int n = n0 + r;
    float xv = x3[((size_t)b * 128 + c) * Npts + n];
    float agg = 0.f;
#pragma unroll
    for (int k = 0; k < KNN; ++k)
      agg += wFin[r][k] * y3T[((size_t)b * Mpts + iFin[r][k]) * 128 + c];
    agg -= xv;
    att[((size_t)b * 256 + c) * Npts + n]       = xv;
    att[((size_t)b * 256 + 128 + c) * Npts + n] = agg;
  }
}

// ------------------------------------------------------------------
// K4: conv (1x1, f16 WMMA) + eval-BatchNorm + LeakyReLU, Cout=256,
// writes into cat channels [256,512).  WG tile: 64(Cout) x 64(N);
// each wave: one A fragment, two B fragments -> 2 WMMAs per staging.
// ------------------------------------------------------------------
__global__ void __launch_bounds__(256)
k_conv_bn(const float* __restrict__ in, const float* __restrict__ W,
          const float* __restrict__ bng, const float* __restrict__ bnb,
          const float* __restrict__ bnm, const float* __restrict__ bnv,
          float* __restrict__ cat, int Cin, int Npts) {
  const int b = blockIdx.z;
  const int cout0 = blockIdx.y * 64;
  const int n0 = blockIdx.x * 64;
  const int tid = threadIdx.x;
  const int wave = tid >> 5, lane = tid & 31;
  const int lmod = lane & 15, lhi = lane >> 4;
  const int wr = wave >> 1, wc = wave & 1;

  __shared__ _Float16 WL[64][40];
  __shared__ _Float16 inL[64][40];

  v8f acc0 = {}, acc1 = {};
  for (int k0 = 0; k0 < Cin; k0 += 32) {
    __syncthreads();
    {
      const int kw = tid & 31, qw = tid >> 5;
#pragma unroll
      for (int i = 0; i < 8; ++i)
        WL[qw + 8 * i][kw] = (_Float16)W[(size_t)(cout0 + qw + 8 * i) * Cin + k0 + kw];
      const int nn = tid & 63, kq = tid >> 6;
#pragma unroll
      for (int i = 0; i < 8; ++i)
        inL[nn][kq + 4 * i] = (_Float16)in[((size_t)b * Cin + k0 + kq + 4 * i) * Npts + n0 + nn];
    }
    __syncthreads();
    v16h af, bf0, bf1;
    const _Float16* wrow  = &WL[wr * 16 + lmod][0];
    const _Float16* n0row = &inL[wc * 32 + lmod][0];
    const _Float16* n1row = &inL[wc * 32 + 16 + lmod][0];
#pragma unroll
    for (int e = 0; e < 8; ++e) af[e] = wrow[16 * lhi + e];
#pragma unroll
    for (int e = 0; e < 8; ++e) af[8 + e] = wrow[16 * lhi + 8 + e];
#pragma unroll
    for (int e = 0; e < 16; ++e) bf0[e] = n0row[16 * lhi + e];
#pragma unroll
    for (int e = 0; e < 16; ++e) bf1[e] = n1row[16 * lhi + e];
    acc0 = __builtin_amdgcn_wmma_f32_16x16x32_f16(false, af, false, bf0,
                                                  (short)0, acc0, false, false);
    acc1 = __builtin_amdgcn_wmma_f32_16x16x32_f16(false, af, false, bf1,
                                                  (short)0, acc1, false, false);
  }
  const int na = n0 + wc * 32 + lmod;
  const int nb = na + 16;
#pragma unroll
  for (int r = 0; r < 8; ++r) {
    const int cout = cout0 + wr * 16 + r + 8 * lhi;
    float sc = bng[cout] * rsqrtf(bnv[cout] + GN_EPS);
    float sh = bnb[cout] - bnm[cout] * sc;
    float v0 = acc0[r] * sc + sh;
    float v1 = acc1[r] * sc + sh;
    v0 = (v0 >= 0.f) ? v0 : 0.2f * v0;
    v1 = (v1 >= 0.f) ? v1 : 0.2f * v1;
    cat[((size_t)b * 512 + 256 + cout) * Npts + na] = v0;
    cat[((size_t)b * 512 + 256 + cout) * Npts + nb] = v1;
  }
}

// ------------------------------------------------------------------
// K5: generic conv (1x1, f16 WMMA) + optional bias; writes pre-GN
// output and accumulates per-(batch,group) sum / sumsq.
// Same 64x64 tile / 2-WMMA-per-wave structure as k_conv_bn.
// ------------------------------------------------------------------
__global__ void __launch_bounds__(256)
k_conv_gn(const float* __restrict__ in, const float* __restrict__ W,
          const float* __restrict__ bias, float* __restrict__ out,
          float* __restrict__ gSum, float* __restrict__ gSqs,
          int Cin, int Cout, int Npts, int chPerGroup) {
  const int b = blockIdx.z;
  const int cout0 = blockIdx.y * 64;
  const int n0 = blockIdx.x * 64;
  const int tid = threadIdx.x;
  const int wave = tid >> 5, lane = tid & 31;
  const int lmod = lane & 15, lhi = lane >> 4;
  const int wr = wave >> 1, wc = wave & 1;

  __shared__ _Float16 WL[64][40];
  __shared__ _Float16 inL[64][40];
  __shared__ float chS[64], chQ[64];
  if (tid < 64) { chS[tid] = 0.f; chQ[tid] = 0.f; }

  v8f acc0 = {}, acc1 = {};
  for (int k0 = 0; k0 < Cin; k0 += 32) {
    __syncthreads();
    {
      const int kw = tid & 31, qw = tid >> 5;
#pragma unroll
      for (int i = 0; i < 8; ++i)
        WL[qw + 8 * i][kw] = (_Float16)W[(size_t)(cout0 + qw + 8 * i) * Cin + k0 + kw];
      const int nn = tid & 63, kq = tid >> 6;
#pragma unroll
      for (int i = 0; i < 8; ++i)
        inL[nn][kq + 4 * i] = (_Float16)in[((size_t)b * Cin + k0 + kq + 4 * i) * Npts + n0 + nn];
    }
    __syncthreads();
    v16h af, bf0, bf1;
    const _Float16* wrow  = &WL[wr * 16 + lmod][0];
    const _Float16* n0row = &inL[wc * 32 + lmod][0];
    const _Float16* n1row = &inL[wc * 32 + 16 + lmod][0];
#pragma unroll
    for (int e = 0; e < 8; ++e) af[e] = wrow[16 * lhi + e];
#pragma unroll
    for (int e = 0; e < 8; ++e) af[8 + e] = wrow[16 * lhi + 8 + e];
#pragma unroll
    for (int e = 0; e < 16; ++e) bf0[e] = n0row[16 * lhi + e];
#pragma unroll
    for (int e = 0; e < 16; ++e) bf1[e] = n1row[16 * lhi + e];
    acc0 = __builtin_amdgcn_wmma_f32_16x16x32_f16(false, af, false, bf0,
                                                  (short)0, acc0, false, false);
    acc1 = __builtin_amdgcn_wmma_f32_16x16x32_f16(false, af, false, bf1,
                                                  (short)0, acc1, false, false);
  }
  __syncthreads();
  const int na = n0 + wc * 32 + lmod;
  const int nb = na + 16;
#pragma unroll
  for (int r = 0; r < 8; ++r) {
    const int co = wr * 16 + r + 8 * lhi;
    const int cout = cout0 + co;
    const float bv = bias ? bias[cout] : 0.f;
    float v0 = acc0[r] + bv;
    float v1 = acc1[r] + bv;
    out[((size_t)b * Cout + cout) * Npts + na] = v0;
    out[((size_t)b * Cout + cout) * Npts + nb] = v1;
    atomicAdd(&chS[co], v0 + v1);
    atomicAdd(&chQ[co], v0 * v0 + v1 * v1);
  }
  __syncthreads();
  if (tid < 64 && (tid % chPerGroup) == 0) {
    float s = 0.f, q = 0.f;
    for (int j = 0; j < chPerGroup; ++j) { s += chS[tid + j]; q += chQ[tid + j]; }
    const int g = (cout0 + tid) / chPerGroup;
    atomicAdd(&gSum[b * 32 + g], s);
    atomicAdd(&gSqs[b * 32 + g], q);
  }
}

// ------------------------------------------------------------------
// K6: GroupNorm apply (+ affine + activation); in-place safe.
// act: 0 = LeakyReLU(0.2), 1 = ReLU
// ------------------------------------------------------------------
__global__ void __launch_bounds__(256)
k_gn_apply(const float* pre, const float* __restrict__ gSum,
           const float* __restrict__ gSqs, const float* __restrict__ gamma,
           const float* __restrict__ beta, float* out,
           int Cout, int Npts, int chPerGroup, int act) {
  size_t idx = (size_t)blockIdx.x * 256 + threadIdx.x;
  int c = (int)((idx / Npts) % Cout);
  int b = (int)(idx / ((size_t)Cout * Npts));
  int g = c / chPerGroup;
  float cnt = (float)chPerGroup * (float)Npts;
  float mu = gSum[b * 32 + g] / cnt;
  float var = gSqs[b * 32 + g] / cnt - mu * mu;
  float v = (pre[idx] - mu) * rsqrtf(fmaxf(var, 0.f) + GN_EPS) * gamma[c] + beta[c];
  v = (act == 0) ? ((v >= 0.f) ? v : 0.2f * v) : fmaxf(v, 0.f);
  out[idx] = v;
}

// ------------------------------------------------------------------
// K7: pack cat channels [0,256) = [x1 | x2 | x3]
// ------------------------------------------------------------------
__global__ void __launch_bounds__(256)
k_pack_cat(const float* __restrict__ x1, const float* __restrict__ x2,
           const float* __restrict__ x3, float* __restrict__ cat, int Npts) {
  size_t idx = (size_t)blockIdx.x * 256 + threadIdx.x;  // over B*256*Npts
  int n = (int)(idx % Npts);
  int c = (int)((idx / Npts) % 256);
  int b = (int)(idx / ((size_t)256 * Npts));
  float v;
  if (c < 64)       v = x1[((size_t)b * 64 + c) * Npts + n];
  else if (c < 128) v = x2[((size_t)b * 64 + (c - 64)) * Npts + n];
  else              v = x3[((size_t)b * 128 + (c - 128)) * Npts + n];
  cat[((size_t)b * 512 + c) * Npts + n] = v;
}

// ------------------------------------------------------------------
// K8: global vector: per (b,c) max and mean of loc over N -> d_out tail
// ------------------------------------------------------------------
__global__ void __launch_bounds__(256)
k_glb(const float* __restrict__ loc, float* __restrict__ outGlb, int Npts) {
  int bc = blockIdx.x;                 // b*512 + c
  int b = bc / 512, c = bc % 512;
  const float* p = loc + (size_t)bc * Npts;
  float mx = -1e30f, sm = 0.f;
  for (int i = threadIdx.x; i < Npts; i += 256) {
    float v = p[i]; mx = fmaxf(mx, v); sm += v;
  }
  __shared__ float sMx[256], sSm[256];
  sMx[threadIdx.x] = mx; sSm[threadIdx.x] = sm;
  __syncthreads();
  for (int s = 128; s > 0; s >>= 1) {
    if (threadIdx.x < s) {
      sMx[threadIdx.x] = fmaxf(sMx[threadIdx.x], sMx[threadIdx.x + s]);
      sSm[threadIdx.x] += sSm[threadIdx.x + s];
    }
    __syncthreads();
  }
  if (threadIdx.x == 0) {
    outGlb[b * 1024 + c]       = sMx[0];
    outGlb[b * 1024 + 512 + c] = sSm[0] / (float)Npts;
  }
}

// ------------------------------------------------------------------
extern "C" void kernel_launch(void* const* d_in, const int* in_sizes, int n_in,
                              void* d_out, int out_size, void* d_ws, size_t ws_size,
                              hipStream_t stream) {
  const int Bb = 4, Npts = 4096, Mpts = 4096;
  const float* x1 = (const float*)d_in[0];
  const float* x2 = (const float*)d_in[1];
  const float* x3 = (const float*)d_in[2];
  const float* y3 = (const float*)d_in[3];
  const float* W4 = (const float*)d_in[4];
  const float* bn4_g = (const float*)d_in[5];
  const float* bn4_b = (const float*)d_in[6];
  const float* bn4_m = (const float*)d_in[7];
  const float* bn4_v = (const float*)d_in[8];
  const float* W5 = (const float*)d_in[9];
  const float* gn5_g = (const float*)d_in[10];
  const float* gn5_b = (const float*)d_in[11];
  const float* W6 = (const float*)d_in[12];
  const float* b6 = (const float*)d_in[13];
  const float* gn6_g = (const float*)d_in[14];
  const float* gn6_b = (const float*)d_in[15];
  const float* W7 = (const float*)d_in[16];
  const float* b7 = (const float*)d_in[17];
  const float* gn7_g = (const float*)d_in[18];
  const float* gn7_b = (const float*)d_in[19];
  const float* W8 = (const float*)d_in[20];
  const float* b8 = (const float*)d_in[21];
  const float* gn8_g = (const float*)d_in[22];
  const float* gn8_b = (const float*)d_in[23];

  float* w = (float*)d_ws;
  float* xnT = w;                 // 2M floats
  float* ynT = w + 2097152;       // 2M
  float* y3T = w + 4194304;       // 2M
  float* att = w + 6291456;       // 4M
  float* cat = w + 10485760;      // 8M
  float* loc = w + 18874368;      // 8M
  float* h6  = w;                 // reuse xnT..att (dead after conv4)
  float* h7  = w + 10485760;      // reuse cat (dead after conv5)
  float* stats = w + 27262976;    // 1024 floats
  float* s5S = stats,       *s5Q = stats + 128;
  float* s6S = stats + 256, *s6Q = stats + 384;
  float* s7S = stats + 512, *s7Q = stats + 640;
  float* s8S = stats + 768, *s8Q = stats + 896;
  float* outF = (float*)d_out;              // [0,2097152) emb, then glb

  (void)hipMemsetAsync(stats, 0, 1024 * sizeof(float), stream);

  k_norm_transpose<<<(Bb * Npts) / 128, 128, 0, stream>>>(x3, xnT, nullptr, Npts);
  k_norm_transpose<<<(Bb * Mpts) / 128, 128, 0, stream>>>(y3, ynT, y3T, Mpts);

  k_simi_att<<<dim3(Npts / 16, Bb), 256, 0, stream>>>(xnT, ynT, y3T, x3, att, Npts, Mpts);

  k_pack_cat<<<(Bb * 256 * Npts) / 256, 256, 0, stream>>>(x1, x2, x3, cat, Npts);
  k_conv_bn<<<dim3(Npts / 64, 4, Bb), 256, 0, stream>>>(att, W4, bn4_g, bn4_b, bn4_m,
                                                        bn4_v, cat, 256, Npts);

  k_conv_gn<<<dim3(Npts / 64, 8, Bb), 256, 0, stream>>>(cat, W5, nullptr, loc, s5S, s5Q,
                                                        512, 512, Npts, 16);
  k_gn_apply<<<(Bb * 512 * Npts) / 256, 256, 0, stream>>>(loc, s5S, s5Q, gn5_g, gn5_b,
                                                          loc, 512, Npts, 16, 0);
  k_glb<<<Bb * 512, 256, 0, stream>>>(loc, outF + 2097152, Npts);

  k_conv_gn<<<dim3(Npts / 64, 8, Bb), 256, 0, stream>>>(loc, W6, b6, h6, s6S, s6Q,
                                                        512, 512, Npts, 16);
  k_gn_apply<<<(Bb * 512 * Npts) / 256, 256, 0, stream>>>(h6, s6S, s6Q, gn6_g, gn6_b,
                                                          h6, 512, Npts, 16, 1);

  k_conv_gn<<<dim3(Npts / 64, 4, Bb), 256, 0, stream>>>(h6, W7, b7, h7, s7S, s7Q,
                                                        512, 256, Npts, 8);
  k_gn_apply<<<(Bb * 256 * Npts) / 256, 256, 0, stream>>>(h7, s7S, s7Q, gn7_g, gn7_b,
                                                          h7, 256, Npts, 8, 1);

  k_conv_gn<<<dim3(Npts / 64, 2, Bb), 256, 0, stream>>>(h7, W8, b8, outF, s8S, s8Q,
                                                        256, 128, Npts, 4);
  k_gn_apply<<<(Bb * 128 * Npts) / 256, 256, 0, stream>>>(outF, s8S, s8Q, gn8_g, gn8_b,
                                                          outF, 128, Npts, 4, 1);
}
